// Eyettention_7490422964772
// MI455X (gfx1250) — matrix operand order repaired and checked by reading, not compiled, not run on hardware
//
#include <hip/hip_runtime.h>
#include <cstdint>
#include <cstddef>

typedef __bf16 bf16_t;
typedef __attribute__((ext_vector_type(16))) __bf16 v16bf;
typedef __attribute__((ext_vector_type(8)))  float  v8f;
typedef __attribute__((ext_vector_type(4)))  unsigned v4u;   // 128-bit chunk (builtin type:
                                                             // assigns across address spaces)

#define LDS_ __attribute__((address_space(3)))

namespace {
constexpr int kB   = 128;   // batch
constexpr int kSN  = 64;    // sentence words
constexpr int kSP  = 128;   // scanpath len
constexpr int kSUB = 96;    // subword len
constexpr int kD   = 768;   // bert dim
constexpr int kH   = 128;   // decoder hidden
constexpr int kHD  = 64;    // encoder hidden per dir
constexpr int kT   = 127;   // decoder steps (SP-1)
constexpr int kOUT = 125;   // SN*2-3
constexpr float kNEG = -1.0e9f;
}

__device__ __forceinline__ float sigf(float x) { return 1.0f / (1.0f + __expf(-x)); }

// ---- WMMA fragment loads (bf16 16x16x32, k = 16*half + s in both A and B) ----
// All call sites guarantee 16B alignment: leading dims are multiples of 16 elements,
// k0 multiples of 32, buffers 256B-aligned.
// A fragment: per-lane elements 0..7 at p[0..7], 8..15 at p[8..15]  (p includes the
// lane-half K offset of +16). Two b128 loads.
__device__ __forceinline__ v16bf load_fragA(const bf16_t* p) {
  union { v4u u[2]; v16bf v; } t;
  t.u[0] = *(const v4u*)(p);
  t.u[1] = *(const v4u*)(p + 8);
  return t.v;
}
__device__ __forceinline__ v16bf load_fragA(const LDS_ bf16_t* p) {
  union { v4u u[2]; v16bf v; } t;
  t.u[0] = *(const LDS_ v4u*)(p);
  t.u[1] = *(const LDS_ v4u*)(p + 8);
  return t.v;
}
// B fragment (from row-major N x K weight, i.e. B = W^T): 16 contiguous bf16 per lane
// (p includes the lane-half K offset of +16). Two b128 loads.
__device__ __forceinline__ v16bf load_fragB(const bf16_t* p) {
  union { v4u u[2]; v16bf v; } t;
  t.u[0] = *(const v4u*)(p);
  t.u[1] = *(const v4u*)(p + 8);
  return t.v;
}
__device__ __forceinline__ v16bf load_fragB(const LDS_ bf16_t* p) {
  union { v4u u[2]; v16bf v; } t;
  t.u[0] = *(const LDS_ v4u*)(p);
  t.u[1] = *(const LDS_ v4u*)(p + 8);
  return t.v;
}

// Strip-mined tile GEMM: each wave computes a 16x(16*STRIP) strip so one A fragment feeds
// STRIP WMMAs. Per-lane base pointers are hoisted out of the K loop; the steady-state loop
// preloads the next A fragment unconditionally and the final K iteration is peeled.
// Pointer types are template parameters so LDS operands stay in address space 3
// (ds_load_b128 / 32-bit addressing) while global operands use global_load_b128.
// Requires (N/16) % STRIP == 0.
// C = A1*W1^T (+ A2*W2^T) (+Cin) (+bias0+bias1), optional relu, f32 and/or bf16 outputs.
template <int STRIP, typename AP, typename WP, typename CFP>
__device__ __forceinline__ void gemm_tiles_dev(
                               AP A1, int lda1, WP W1, int ldk1, int K1,
                               const bf16_t* A2, int lda2, const bf16_t* W2, int ldk2, int K2,
                               const float* bias0, const float* bias1,
                               const float* Cin, int ldcin,
                               CFP Cf, bf16_t* Cb, int ldc,
                               int M, int N, int Nreal, int relu,
                               int gw, int nw, int lane) {
  int mt = M >> 4, nt = N >> 4;
  int nstrips = nt / STRIP;
  int total = mt * nstrips;
  int half = lane >> 4, ln = lane & 15;
  int aoff = (lane & 16) ? 16 : 0;
  int boff = (lane & 16) ? 16 : 0;
  for (int s = gw; s < total; s += nw) {
    int m0 = (s / nstrips) << 4;
    int n0 = (s % nstrips) * (STRIP * 16);
    v8f acc[STRIP];
    // init: Cin + guarded biases
#pragma unroll
    for (int u = 0; u < STRIP; ++u) {
      int n = n0 + u * 16 + ln;
      float bv = 0.0f;
      if (n < Nreal) {
        if (bias0) bv += bias0[n];
        if (bias1) bv += bias1[n];
      }
      const float* cin = Cin ? (Cin + (size_t)(m0 + half * 8) * ldcin + n) : nullptr;
#pragma unroll
      for (int r = 0; r < 8; ++r)
        acc[u][r] = bv + (cin ? cin[(size_t)r * ldcin] : 0.0f);
    }
    if (K1 > 0) {
      AP ap = A1 + (size_t)(m0 + ln) * lda1 + aoff;
      WP bp[STRIP];
#pragma unroll
      for (int u = 0; u < STRIP; ++u) bp[u] = W1 + (size_t)(n0 + u * 16 + ln) * ldk1 + boff;
      v16bf a = load_fragA(ap);
      int k0 = 0;
      for (; k0 + 32 < K1; k0 += 32) {           // steady state: unconditional preload
        v16bf an = load_fragA(ap + k0 + 32);
#pragma unroll
        for (int u = 0; u < STRIP; ++u) {
          v16bf b = load_fragB(bp[u] + k0);
          acc[u] = __builtin_amdgcn_wmma_f32_16x16x32_bf16(false, a, false, b, (short)0,
                                                           acc[u], false, false);
        }
        a = an;
      }
#pragma unroll
      for (int u = 0; u < STRIP; ++u) {          // peeled final K iteration
        v16bf b = load_fragB(bp[u] + k0);
        acc[u] = __builtin_amdgcn_wmma_f32_16x16x32_bf16(false, a, false, b, (short)0,
                                                         acc[u], false, false);
      }
    }
    if (A2 && K2 > 0) {
      const bf16_t* ap = A2 + (size_t)(m0 + ln) * lda2 + aoff;
      const bf16_t* bp[STRIP];
#pragma unroll
      for (int u = 0; u < STRIP; ++u) bp[u] = W2 + (size_t)(n0 + u * 16 + ln) * ldk2 + boff;
      v16bf a = load_fragA(ap);
      int k0 = 0;
      for (; k0 + 32 < K2; k0 += 32) {
        v16bf an = load_fragA(ap + k0 + 32);
#pragma unroll
        for (int u = 0; u < STRIP; ++u) {
          v16bf b = load_fragB(bp[u] + k0);
          acc[u] = __builtin_amdgcn_wmma_f32_16x16x32_bf16(false, a, false, b, (short)0,
                                                           acc[u], false, false);
        }
        a = an;
      }
#pragma unroll
      for (int u = 0; u < STRIP; ++u) {
        v16bf b = load_fragB(bp[u] + k0);
        acc[u] = __builtin_amdgcn_wmma_f32_16x16x32_bf16(false, a, false, b, (short)0,
                                                         acc[u], false, false);
      }
    }
    // epilogue: uniform branch keeps the common (no column padding) path compare-free
    if (Nreal >= N) {
#pragma unroll
      for (int u = 0; u < STRIP; ++u) {
        size_t o = (size_t)(m0 + half * 8) * ldc + n0 + u * 16 + ln;
#pragma unroll
        for (int r = 0; r < 8; ++r) {
          float v = acc[u][r];
          if (relu) v = v > 0.0f ? v : 0.0f;
          if (Cf) Cf[o + (size_t)r * ldc] = v;
          if (Cb) Cb[o + (size_t)r * ldc] = (bf16_t)v;
        }
      }
    } else {
#pragma unroll
      for (int u = 0; u < STRIP; ++u) {
        int n = n0 + u * 16 + ln;
        size_t o = (size_t)(m0 + half * 8) * ldc + n;
#pragma unroll
        for (int r = 0; r < 8; ++r) {
          float v = acc[u][r];
          if (relu) v = v > 0.0f ? v : 0.0f;
          if (Cf && n < Nreal) Cf[o + (size_t)r * ldc] = v;
          if (Cb) Cb[o + (size_t)r * ldc] = (bf16_t)v;
        }
      }
    }
  }
}

__global__ void k_gemm(const bf16_t* A, int lda, const bf16_t* W, int ldk, int K,
                       const float* bias0, const float* bias1,
                       const float* Cin, int ldcin,
                       float* Cf, bf16_t* Cb, int ldc,
                       int M, int N, int Nreal, int relu) {
  int gw = blockIdx.x * (blockDim.x >> 5) + (threadIdx.x >> 5);
  int nw = gridDim.x * (blockDim.x >> 5);
  gemm_tiles_dev<4>(A, lda, W, ldk, K, nullptr, 0, nullptr, 0, 0,
                    bias0, bias1, Cin, ldcin, Cf, Cb, ldc, M, N, Nreal, relu,
                    gw, nw, threadIdx.x & 31);
}

// ---- small prep kernels ----
__global__ void k_pack(const float* W, bf16_t* Wb, int O, int I, int Ipad, int total) {
  int idx = blockIdx.x * 256 + threadIdx.x;
  if (idx >= total) return;
  int o = idx / Ipad, i = idx % Ipad;
  Wb[idx] = (o < O && i < I) ? (bf16_t)W[(size_t)o * I + i] : (bf16_t)0.0f;
}

__global__ void k_pool_sn(const float* hid, const int* ids, float* merged) {
  int b = blockIdx.x;
  for (int d = threadIdx.x; d < kD; d += blockDim.x)
    for (int l = 0; l < kSUB; ++l) {
      int w = ids[b * kSUB + l];
      merged[((size_t)b * kSN + w) * kD + d] += hid[((size_t)b * kSUB + l) * kD + d];
    }
}

__global__ void k_mask(const float* merged, int* bad) {
  int idx = blockIdx.x * 256 + threadIdx.x;
  if (idx >= kB * kSN) return;
  float s = 0.0f;
  for (int d = 0; d < kD; ++d) s += merged[(size_t)idx * kD + d];
  bad[idx] = (s == 0.0f) ? 1 : 0;
}

__global__ void k_x0(const float* merged, bf16_t* X0) {
  size_t idx = (size_t)blockIdx.x * 256 + threadIdx.x;
  if (idx >= (size_t)kSN * kB * kD) return;
  int d = (int)(idx % kD);
  size_t tb = idx / kD;
  int b = (int)(tb % kB), t = (int)(tb / kB);
  X0[idx] = (bf16_t)merged[((size_t)b * kSN + t) * kD + d];
}

__global__ void k_encout(const float* Xf, const float* wlen, float* enc_out) {
  int idx = blockIdx.x * 256 + threadIdx.x;
  if (idx >= kB * kSN * 129) return;
  int d = idx % 129, bs = idx / 129;
  int b = bs >> 6, s = bs & 63;
  enc_out[idx] = (d < kH) ? Xf[((size_t)s * kB + b) * kH + d] : wlen[b * kSN + s];
}

__global__ void k_pool_sp(const float* wemb, const int* sp_emd, const int* wid_sp, float* sp_mrg) {
  int b = blockIdx.x;
  for (int d = threadIdx.x; d < kD; d += blockDim.x)
    for (int l = 0; l < kT; ++l) {
      int w = wid_sp[b * kSP + l];
      sp_mrg[((size_t)b * kT + w) * kD + d] += wemb[(size_t)sp_emd[b * kSP + l] * kD + d];
    }
}

__global__ void k_dec_ln(const float* sp_mrg, const float* pemb, const int* sp_pos,
                         const float* lng, const float* lnb,
                         const float* land, const float* fixd, bf16_t* dec_in) {
  __shared__ float vals[kD];
  __shared__ float red[256];
  int row = blockIdx.x;               // b*kT + w
  int b = row / kT, w = row % kT;
  const float* pe = pemb + (size_t)sp_pos[b * kSP + w] * kD;
  float ls = 0.0f, lq = 0.0f;
  for (int d = threadIdx.x; d < kD; d += 256) {
    float v = sp_mrg[(size_t)row * kD + d] + pe[d];
    vals[d] = v; ls += v; lq += v * v;
  }
  red[threadIdx.x] = ls; __syncthreads();
  for (int o = 128; o > 0; o >>= 1) { if (threadIdx.x < o) red[threadIdx.x] += red[threadIdx.x + o]; __syncthreads(); }
  float mu = red[0] / (float)kD; __syncthreads();
  red[threadIdx.x] = lq; __syncthreads();
  for (int o = 128; o > 0; o >>= 1) { if (threadIdx.x < o) red[threadIdx.x] += red[threadIdx.x + o]; __syncthreads(); }
  float var = red[0] / (float)kD - mu * mu;
  float inv = rsqrtf(var + 1e-12f);
  bf16_t* out = dec_in + ((size_t)w * kB + b) * 800;
  for (int d = threadIdx.x; d < kD; d += 256)
    out[d] = (bf16_t)((vals[d] - mu) * inv * lng[d] + lnb[d]);
  if (threadIdx.x == 0) {
    out[768] = (bf16_t)land[b * kSP + w];
    out[769] = (bf16_t)fixd[b * kSP + w];
  }
}

// ---- encoder recurrent scan: 1 block per direction, state AND Whh in LDS (AS3-typed so
// the GEMM operand traffic is pure ds_load_b128), WMMA per step ----
__global__ void __launch_bounds__(1024) k_enc_scan(const float* G0, const float* G1,
                                                   const bf16_t* Whh0, const bf16_t* Whh1,
                                                   bf16_t* Xnext, float* Xf) {
  extern __shared__ char smem_raw[];
  LDS_ char* sm = (LDS_ char*)smem_raw;
  LDS_ bf16_t* sh_hbf  = (LDS_ bf16_t*)sm;                                   // 128*64 bf16 (16KB)
  LDS_ bf16_t* sh_whh  = (LDS_ bf16_t*)(sm + kB * kHD * 2);                  // 256*64 bf16 (32KB)
  LDS_ float*  sh_gate = (LDS_ float*)(sm + kB * kHD * 2 + 256 * kHD * 2);   // 128*256 f32 (128KB)
  LDS_ float*  sh_c    = (LDS_ float*)(sm + kB * kHD * 2 + 256 * kHD * 2 + kB * 256 * 4); // 32KB
  int dir = blockIdx.x;
  const float* G = dir ? G1 : G0;
  const bf16_t* Whh = dir ? Whh1 : Whh0;
  int lane = threadIdx.x & 31, wv = threadIdx.x >> 5;
  for (int i = threadIdx.x; i < kB * kHD; i += 1024) { sh_c[i] = 0.0f; sh_hbf[i] = (bf16_t)0.0f; }
  for (int i = threadIdx.x; i < 256 * kHD; i += 1024) sh_whh[i] = Whh[i];  // stage Whh once
  __syncthreads();
  for (int it = 0; it < kSN; ++it) {
    int t = dir ? (kSN - 1 - it) : it;
    // gates = G[t] + h @ Whh^T   (32 strips of 16x64 == 32 waves)
    gemm_tiles_dev<4>((LDS_ const bf16_t*)sh_hbf, kHD, (LDS_ const bf16_t*)sh_whh, kHD, kHD,
                      nullptr, 0, nullptr, 0, 0,
                      nullptr, nullptr,
                      G + (size_t)t * kB * 256, 256,
                      sh_gate, (bf16_t*)nullptr, 256, kB, 256, 256, 0, wv, 32, lane);
    __syncthreads();
    for (int idx = threadIdx.x; idx < kB * kHD; idx += 1024) {
      int b = idx >> 6, hd = idx & 63;
      LDS_ const float* g = sh_gate + (size_t)b * 256;
      float cc = sigf(g[kHD + hd]) * sh_c[idx] + sigf(g[hd]) * tanhf(g[2 * kHD + hd]);
      float hh = sigf(g[3 * kHD + hd]) * tanhf(cc);
      sh_c[idx] = cc;
      sh_hbf[idx] = (bf16_t)hh;
      size_t o = ((size_t)t * kB + b) * kH + dir * kHD + hd;
      Xnext[o] = (bf16_t)hh;
      Xf[o] = hh;
    }
    __syncthreads();
  }
}

// ---- persistent decoder ----
struct DecP {
  const float* G0;
  const bf16_t* Wih[8];
  const bf16_t* Whh[8];
  const float* bih[8];
  const float* bhh[8];
  const bf16_t* attnW;
  const float* attnb;
  const bf16_t* dW[5];
  const float* db[5];
  const float* enc_out;
  const int* bad;
  const int* sp_pos;
  float* gates;
  float* h;
  float* c;
  bf16_t* hbf;
  float* q;
  bf16_t* z0; bf16_t* z1; bf16_t* z2; bf16_t* z3; bf16_t* z4;
  float* outs;
  float* aws;
  unsigned* bar;
};

__device__ void gbar(unsigned* bar) {
  __syncthreads();
  if (threadIdx.x == 0) {
    __threadfence();
    volatile unsigned* vgen = bar + 1;
    unsigned g = *vgen;
    if (atomicAdd(bar, 1u) == gridDim.x - 1) {
      *bar = 0u;
      __threadfence();
      atomicAdd(bar + 1, 1u);
    } else {
      while (*vgen == g) __builtin_amdgcn_s_sleep(2);
    }
    __threadfence();
  }
  __syncthreads();
}

__global__ void __launch_bounds__(1024) k_dec_main(DecP p) {
  int gw = blockIdx.x * 32 + (threadIdx.x >> 5);
  int nw = gridDim.x * 32;
  int lane = threadIdx.x & 31;
  int tid = blockIdx.x * 1024 + threadIdx.x;
  int nthr = gridDim.x * 1024;
  for (int t = 0; t < kT; ++t) {
    // ---- 8 stacked LSTM layers (Wih and Whh fused into one accumulator pass) ----
    for (int l = 0; l < 8; ++l) {
      const bf16_t* hin = p.hbf + (size_t)l * kB * kH;
      const bf16_t* xin = (l == 0) ? nullptr : (p.hbf + (size_t)(l - 1) * kB * kH);
      gemm_tiles_dev<1>(hin, kH, p.Whh[l], kH, kH,
                        xin, kH, p.Wih[l], kH, (l == 0) ? 0 : kH,
                        (l == 0) ? nullptr : p.bih[l], (l == 0) ? nullptr : p.bhh[l],
                        (l == 0) ? (p.G0 + (size_t)t * kB * 512) : nullptr, 512,
                        p.gates, nullptr, 512, kB, 512, 512, 0, gw, nw, lane);
      gbar(p.bar);
      for (int idx = tid; idx < kB * kH; idx += nthr) {
        int b = idx >> 7, j = idx & (kH - 1);
        const float* g = p.gates + (size_t)b * 512;
        size_t si = (size_t)l * kB * kH + idx;
        float cc = sigf(g[kH + j]) * p.c[si] + sigf(g[j]) * tanhf(g[2 * kH + j]);
        float hh = sigf(g[3 * kH + j]) * tanhf(cc);
        p.c[si] = cc;
        p.h[si] = hh;
        p.hbf[si] = (bf16_t)hh;
      }
      gbar(p.bar);
    }
    // ---- q = h7 @ attnW^T + attn_b ----
    gemm_tiles_dev<1>(p.hbf + (size_t)7 * kB * kH, kH, p.attnW, kH, kH,
                      nullptr, 0, nullptr, 0, 0,
                      p.attnb, nullptr, nullptr, 0,
                      p.q, nullptr, 144, kB, 144, 129, 0, gw, nw, lane);
    gbar(p.bar);
    // ---- attention: one wave per batch row ----
    if (gw < kB) {
      int b = gw;
      int pos = p.sp_pos[b * kSP + t];
      int left = pos - 1 > 0 ? pos - 1 : 0;
      int right = pos + 1 < kSN - 1 ? pos + 1 : kSN - 1;
      const float* qp = p.q + (size_t)b * 144;
      float sc[2];
#pragma unroll
      for (int u = 0; u < 2; ++u) {
        int s = lane + u * 32;
        const float* ep = p.enc_out + ((size_t)b * kSN + s) * 129;
        float acc = 0.0f;
        for (int d = 0; d < 129; ++d) acc += qp[d] * ep[d];
        bool msk = (p.bad[b * kSN + s] != 0) || (s < left) || (s > right);
        sc[u] = acc + (msk ? kNEG : 0.0f);
      }
      float mx = fmaxf(sc[0], sc[1]);
      for (int o = 16; o > 0; o >>= 1) mx = fmaxf(mx, __shfl_xor(mx, o, 32));
      float e0 = __expf(sc[0] - mx), e1 = __expf(sc[1] - mx);
      float sm = e0 + e1;
      for (int o = 16; o > 0; o >>= 1) sm += __shfl_xor(sm, o, 32);
      float inv = 1.0f / sm;
      float dp0 = (float)(lane - pos), dp1 = (float)(lane + 32 - pos);
      float a0 = e0 * inv * __expf(-2.0f * dp0 * dp0);   // SIG=0.5 -> 1/(2s^2)=2
      float a1 = e1 * inv * __expf(-2.0f * dp1 * dp1);
      p.aws[((size_t)b * kT + t) * kSN + lane] = a0;
      p.aws[((size_t)b * kT + t) * kSN + lane + 32] = a1;
      for (int d = lane; d < 129; d += 32) {
        float acc = 0.0f;
        for (int s = 0; s < kSN; ++s) {
          float a = (s < 32) ? __shfl(a0, s, 32) : __shfl(a1, s - 32, 32);
          acc += a * p.enc_out[((size_t)b * kSN + s) * 129 + d];
        }
        p.z0[(size_t)b * 288 + d] = (bf16_t)acc;
      }
      for (int j = lane; j < kH; j += 32)
        p.z0[(size_t)b * 288 + 129 + j] = p.hbf[(size_t)7 * kB * kH + b * kH + j];
      for (int j = lane; j < 31; j += 32)
        p.z0[(size_t)b * 288 + 257 + j] = (bf16_t)0.0f;
    }
    gbar(p.bar);
    // ---- MLP head ----
    gemm_tiles_dev<1>(p.z0, 288, p.dW[0], 288, 288, nullptr, 0, nullptr, 0, 0,
                      p.db[0], nullptr, nullptr, 0, (float*)nullptr, p.z1, 512, kB, 512, 512, 1, gw, nw, lane);
    gbar(p.bar);
    gemm_tiles_dev<1>(p.z1, 512, p.dW[1], 512, 512, nullptr, 0, nullptr, 0, 0,
                      p.db[1], nullptr, nullptr, 0, (float*)nullptr, p.z2, 256, kB, 256, 256, 1, gw, nw, lane);
    gbar(p.bar);
    gemm_tiles_dev<1>(p.z2, 256, p.dW[2], 256, 256, nullptr, 0, nullptr, 0, 0,
                      p.db[2], nullptr, nullptr, 0, (float*)nullptr, p.z3, 256, kB, 256, 256, 1, gw, nw, lane);
    gbar(p.bar);
    gemm_tiles_dev<1>(p.z3, 256, p.dW[3], 256, 256, nullptr, 0, nullptr, 0, 0,
                      p.db[3], nullptr, nullptr, 0, (float*)nullptr, p.z4, 256, kB, 256, 256, 1, gw, nw, lane);
    gbar(p.bar);
    gemm_tiles_dev<1>(p.z4, 256, p.dW[4], 256, 256, nullptr, 0, nullptr, 0, 0,
                      p.db[4], nullptr, nullptr, 0,
                      p.outs + (size_t)t * kOUT, nullptr, kT * kOUT, kB, 128, kOUT, 0, gw, nw, lane);
    gbar(p.bar);
  }
}

// ---- host ----
static void launch_gemm(hipStream_t s, const bf16_t* A, int lda, const bf16_t* W, int ldk, int K,
                        const float* b0, const float* b1, const float* Cin, int ldcin,
                        float* Cf, bf16_t* Cb, int ldc, int M, int N, int Nreal, int relu) {
  if (Nreal == 0) Nreal = N;
  int strips = (M / 16) * ((N / 16) / 4);   // STRIP=4; (N/16) % 4 == 0 at all call sites
  int blocks = (strips + 7) / 8;
  if (blocks > 4096) blocks = 4096;
  if (blocks < 1) blocks = 1;
  k_gemm<<<blocks, 256, 0, s>>>(A, lda, W, ldk, K, b0, b1, Cin, ldcin, Cf, Cb, ldc, M, N, Nreal, relu);
}

extern "C" void kernel_launch(void* const* d_in, const int* in_sizes, int n_in,
                              void* d_out, int out_size, void* d_ws, size_t ws_size,
                              hipStream_t stream) {
  (void)in_sizes; (void)n_in; (void)out_size; (void)ws_size;
  const float* sn_bert = (const float*)d_in[0];
  const int*   wid_sn  = (const int*)d_in[1];
  const float* wlen    = (const float*)d_in[2];
  const int*   sp_emd  = (const int*)d_in[3];
  const int*   sp_pos  = (const int*)d_in[4];
  const int*   wid_sp  = (const int*)d_in[5];
  const float* fixd    = (const float*)d_in[6];
  const float* land    = (const float*)d_in[7];
  const float* wemb    = (const float*)d_in[8];
  const float* pemb    = (const float*)d_in[9];
  const float* lng     = (const float*)d_in[10];
  const float* lnb     = (const float*)d_in[11];
  // enc_lstm: d_in[12 + (l*2+dir)*4 + {Wih,Whh,bih,bhh}], dec_cells: 76 + l*4, attn 108/109, dense 110+2k

  char* base = (char*)d_ws;
  size_t off = 0;
  auto AL = [&](size_t bytes) -> char* {
    off = (off + 255) & ~(size_t)255;
    char* p = base + off;
    off += bytes;
    return p;
  };
  float*  merged  = (float*) AL((size_t)kB * kSN * kD * 4);
  int*    bad     = (int*)   AL((size_t)kB * kSN * 4);
  bf16_t* X0      = (bf16_t*)AL((size_t)kSN * kB * kD * 2);
  bf16_t* Xa      = (bf16_t*)AL((size_t)kSN * kB * kH * 2);
  bf16_t* Xb      = (bf16_t*)AL((size_t)kSN * kB * kH * 2);
  float*  Xf      = (float*) AL((size_t)kSN * kB * kH * 4);
  float*  Ge0     = (float*) AL((size_t)kSN * kB * 256 * 4);
  float*  Ge1     = (float*) AL((size_t)kSN * kB * 256 * 4);
  float*  enc_out = (float*) AL((size_t)kB * kSN * 129 * 4);
  float*  sp_mrg  = (float*) AL((size_t)kB * kT * kD * 4);
  bf16_t* dec_in  = (bf16_t*)AL((size_t)kT * kB * 800 * 2);
  float*  G0d     = (float*) AL((size_t)kT * kB * 512 * 4);
  float*  hbuf    = (float*) AL((size_t)8 * kB * kH * 4);
  float*  cbuf    = (float*) AL((size_t)8 * kB * kH * 4);
  bf16_t* hbf     = (bf16_t*)AL((size_t)8 * kB * kH * 2);
  float*  gates   = (float*) AL((size_t)kB * 512 * 4);
  float*  qbuf    = (float*) AL((size_t)kB * 144 * 4);
  bf16_t* z0      = (bf16_t*)AL((size_t)kB * 288 * 2);
  bf16_t* z1      = (bf16_t*)AL((size_t)kB * 512 * 2);
  bf16_t* z2      = (bf16_t*)AL((size_t)kB * 256 * 2);
  bf16_t* z3      = (bf16_t*)AL((size_t)kB * 256 * 2);
  bf16_t* z4      = (bf16_t*)AL((size_t)kB * 256 * 2);
  unsigned* bar   = (unsigned*)AL(256);
  bf16_t* encWih[8][2]; bf16_t* encWhh[8][2];
  for (int l = 0; l < 8; ++l)
    for (int d = 0; d < 2; ++d) {
      int din = (l == 0) ? kD : kH;
      encWih[l][d] = (bf16_t*)AL((size_t)256 * din * 2);
      encWhh[l][d] = (bf16_t*)AL((size_t)256 * 64 * 2);
    }
  bf16_t* decWih[8]; bf16_t* decWhh[8];
  for (int l = 0; l < 8; ++l) {
    decWih[l] = (bf16_t*)AL((size_t)512 * ((l == 0) ? 800 : 128) * 2);
    decWhh[l] = (bf16_t*)AL((size_t)512 * 128 * 2);
  }
  bf16_t* attnWb = (bf16_t*)AL((size_t)144 * 128 * 2);
  const int dI[5]  = {257, 512, 256, 256, 256};
  const int dIp[5] = {288, 512, 256, 256, 256};
  const int dO[5]  = {512, 256, 256, 256, 125};
  const int dOp[5] = {512, 256, 256, 256, 128};
  bf16_t* denseWb[5];
  for (int k = 0; k < 5; ++k) denseWb[k] = (bf16_t*)AL((size_t)dOp[k] * dIp[k] * 2);

  (void)hipMemsetAsync(merged, 0, (size_t)kB * kSN * kD * 4, stream);
  (void)hipMemsetAsync(sp_mrg, 0, (size_t)kB * kT * kD * 4, stream);
  (void)hipMemsetAsync(dec_in, 0, (size_t)kT * kB * 800 * 2, stream);
  (void)hipMemsetAsync(hbuf, 0, (size_t)8 * kB * kH * 4, stream);
  (void)hipMemsetAsync(cbuf, 0, (size_t)8 * kB * kH * 4, stream);
  (void)hipMemsetAsync(hbf, 0, (size_t)8 * kB * kH * 2, stream);
  (void)hipMemsetAsync(bar, 0, 256, stream);

  auto pack = [&](const void* src, bf16_t* dst, int O, int I, int Ipad, int Opad) {
    int total = Opad * Ipad;
    k_pack<<<(total + 255) / 256, 256, 0, stream>>>((const float*)src, dst, O, I, Ipad, total);
  };
  for (int l = 0; l < 8; ++l)
    for (int d = 0; d < 2; ++d) {
      int wi = 12 + (l * 2 + d) * 4;
      int din = (l == 0) ? kD : kH;
      pack(d_in[wi], encWih[l][d], 256, din, din, 256);
      pack(d_in[wi + 1], encWhh[l][d], 256, 64, 64, 256);
    }
  for (int l = 0; l < 8; ++l) {
    int wi = 76 + l * 4;
    pack(d_in[wi], decWih[l], 512, (l == 0) ? 770 : 128, (l == 0) ? 800 : 128, 512);
    pack(d_in[wi + 1], decWhh[l], 512, 128, 128, 512);
  }
  pack(d_in[108], attnWb, 129, 128, 128, 144);
  for (int k = 0; k < 5; ++k) pack(d_in[110 + 2 * k], denseWb[k], dO[k], dI[k], dIp[k], dOp[k]);

  // encoder front-end
  k_pool_sn<<<kB, 256, 0, stream>>>(sn_bert, wid_sn, merged);
  k_mask<<<(kB * kSN + 255) / 256, 256, 0, stream>>>(merged, bad);
  {
    size_t total = (size_t)kSN * kB * kD;
    k_x0<<<(unsigned)((total + 255) / 256), 256, 0, stream>>>(merged, X0);
  }
  const bf16_t* Xcur = X0;
  int din = kD;
  const unsigned scan_smem = kB * kHD * 2 + 256 * kHD * 2 + kB * 256 * 4 + kB * kHD * 4;
  for (int l = 0; l < 8; ++l) {
    for (int d = 0; d < 2; ++d) {
      int wi = 12 + (l * 2 + d) * 4;
      launch_gemm(stream, Xcur, din, encWih[l][d], din, din,
                  (const float*)d_in[wi + 2], (const float*)d_in[wi + 3],
                  nullptr, 0, (d ? Ge1 : Ge0), nullptr, 256, kSN * kB, 256, 0, 0);
    }
    bf16_t* Xn = (l & 1) ? Xb : Xa;
    k_enc_scan<<<2, 1024, scan_smem, stream>>>(Ge0, Ge1, encWhh[l][0], encWhh[l][1], Xn, Xf);
    Xcur = Xn;
    din = kH;
  }
  k_encout<<<(kB * kSN * 129 + 255) / 256, 256, 0, stream>>>(Xf, wlen, enc_out);

  // decoder front-end
  k_pool_sp<<<kB, 256, 0, stream>>>(wemb, sp_emd, wid_sp, sp_mrg);
  k_dec_ln<<<kB * kT, 256, 0, stream>>>(sp_mrg, pemb, sp_pos, lng, lnb, land, fixd, dec_in);
  launch_gemm(stream, dec_in, 800, decWih[0], 800, 800,
              (const float*)d_in[76 + 2], (const float*)d_in[76 + 3],
              nullptr, 0, G0d, nullptr, 512, kT * kB, 512, 0, 0);

  // persistent decoder
  DecP p{};
  p.G0 = G0d;
  for (int l = 0; l < 8; ++l) {
    p.Wih[l] = decWih[l];
    p.Whh[l] = decWhh[l];
    p.bih[l] = (const float*)d_in[76 + l * 4 + 2];
    p.bhh[l] = (const float*)d_in[76 + l * 4 + 3];
  }
  p.attnW = attnWb;
  p.attnb = (const float*)d_in[109];
  for (int k = 0; k < 5; ++k) {
    p.dW[k] = denseWb[k];
    p.db[k] = (const float*)d_in[110 + 2 * k + 1];
  }
  p.enc_out = enc_out;
  p.bad = bad;
  p.sp_pos = sp_pos;
  p.gates = gates; p.h = hbuf; p.c = cbuf; p.hbf = hbf;
  p.q = qbuf; p.z0 = z0; p.z1 = z1; p.z2 = z2; p.z3 = z3; p.z4 = z4;
  p.outs = (float*)d_out;
  p.aws = (float*)d_out + (size_t)kB * kT * kOUT;
  p.bar = bar;
  k_dec_main<<<8, 1024, 0, stream>>>(p);
}
